// SimpleDiscreteKeyValueBottleneck_14096082666002
// MI455X (gfx1250) — matrix operand order repaired and checked by reading, not compiled, not run on hardware
//
#include <hip/hip_runtime.h>
#include <hip/hip_bf16.h>

// Problem constants (from reference): B,C,K,D,V
#define BQ 2048
#define CQ 32
#define KQ 2048
#define DQ 64
#define VQ 128

#define NT 4              // N-tiles (of 16 keys) staged per LDS buffer
#define ROWB 144          // LDS row stride bytes (128B data + 16B TDM pad -> conflict-free b128)
#define NSTAGES (KQ / (NT * 16))

typedef __bf16 bf16_t;
typedef __attribute__((ext_vector_type(16))) __bf16 v16bf;
typedef __attribute__((ext_vector_type(8)))  __bf16 v8bf;
typedef __attribute__((ext_vector_type(4)))  __bf16 v4bf;
typedef __attribute__((ext_vector_type(8)))  float  v8f;
typedef __attribute__((ext_vector_type(4)))  unsigned int u32x4;
typedef __attribute__((ext_vector_type(8)))  int i32x8;
typedef __attribute__((ext_vector_type(4)))  int i32x4;

union V16U { v16bf v; v8bf h[2]; };

// ---------------------------------------------------------------------------
// Kernel 1: split batch f32 -> bf16 hi + bf16 lo (elementwise, vectorized x4)
// ---------------------------------------------------------------------------
__global__ __launch_bounds__(256)
void convert_batch_kernel(const float* __restrict__ src,
                          bf16_t* __restrict__ hi, bf16_t* __restrict__ lo) {
  size_t i = ((size_t)blockIdx.x * 256 + threadIdx.x) * 4;   // exact multiple, no tail
  float4 x = *(const float4*)(src + i);
  float xs[4] = {x.x, x.y, x.z, x.w};
  v4bf h, l;
#pragma unroll
  for (int j = 0; j < 4; ++j) {
    bf16_t hh = (bf16_t)xs[j];
    h[j] = hh;
    l[j] = (bf16_t)(xs[j] - (float)hh);
  }
  *(v4bf*)(hi + i) = h;
  *(v4bf*)(lo + i) = l;
}

// ---------------------------------------------------------------------------
// Kernel 2: split keys f32 -> bf16 hi/lo and compute k2[c,k] = sum_d keys^2
// ---------------------------------------------------------------------------
__global__ __launch_bounds__(256)
void convert_keys_kernel(const float* __restrict__ src,
                         bf16_t* __restrict__ hi, bf16_t* __restrict__ lo,
                         float* __restrict__ k2) {
  size_t row = (size_t)blockIdx.x * 256 + threadIdx.x;       // 0 .. C*K-1 (exact)
  const float4* s4 = (const float4*)(src + row * DQ);
  bf16_t* dh = hi + row * DQ;
  bf16_t* dl = lo + row * DQ;
  float acc = 0.f;
#pragma unroll 4
  for (int q = 0; q < DQ / 4; ++q) {
    float4 x = s4[q];
    float xs[4] = {x.x, x.y, x.z, x.w};
    v4bf h, l;
#pragma unroll
    for (int j = 0; j < 4; ++j) {
      acc = fmaf(xs[j], xs[j], acc);
      bf16_t hh = (bf16_t)xs[j];
      h[j] = hh;
      l[j] = (bf16_t)(xs[j] - (float)hh);
    }
    *(v4bf*)(dh + q * 4) = h;
    *(v4bf*)(dl + q * 4) = l;
  }
  k2[row] = acc;
}

// ---------------------------------------------------------------------------
// TDM: DMA one 64-row x 64-element (bf16) key tile into LDS, inserting a 16B
// pad after every 128B row -> LDS row stride 144B (16 distinct bank-quads,
// conflict-free ds_load_b128). D# built per CDNA5 ISA ch.8.
// ---------------------------------------------------------------------------
__device__ __forceinline__
void tdm_load_tile(const bf16_t* gsrc, unsigned lds_off) {
  unsigned long long ga = (unsigned long long)(uintptr_t)gsrc;
  u32x4 g0;
  g0[0] = 1u;                                               // count=1, user mode
  g0[1] = lds_off;                                          // D#.lds_addr (bytes)
  g0[2] = (unsigned)(ga & 0xFFFFFFFFu);                     // global_addr[31:0]
  g0[3] = (unsigned)((ga >> 32) & 0x01FFFFFFu) | (2u << 30);// global_addr[56:32], type=2
  i32x8 g1;
  g1[0] = (int)((1u << 16)    // data_size = 1 -> 2 bytes
              | (1u << 20)    // pad_enable
              | (4u << 22)    // pad_interval: pad after every 32 DWORDs (128B)
              | (3u << 25));  // pad_amount:   4 DWORDs (16B)
  g1[1] = (int)(64u << 16);                 // tensor_dim0 = 64 (bits[79:48], low half)
  g1[2] = (int)((2048u & 0xFFFFu) << 16);   // tensor_dim0 hi = 0 ; tensor_dim1 lo16
  g1[3] = (int)((2048u >> 16) | (64u << 16)); // tensor_dim1 hi ; tile_dim0 = 64
  g1[4] = (int)64u;                         // tile_dim1 = 64 ; tile_dim2 = 0
  g1[5] = (int)64u;                         // tensor_dim0_stride = 64 elements (low 32)
  g1[6] = 0;                                // stride hi / dim1_stride lo
  g1[7] = 0;                                // dim1_stride hi
  i32x4 z4 = {0, 0, 0, 0};                  // groups 2/3 unused (2D tile)
  i32x8 z8 = {0, 0, 0, 0, 0, 0, 0, 0};      // extra descriptor group (unused)
  __builtin_amdgcn_tensor_load_to_lds(g0, g1, z4, z4, z8, 0);
}

// ---------------------------------------------------------------------------
// Kernel 3: per-codebook GEMM (split-bf16 WMMA) fused with argmin epilogue.
// Block = 256 threads = 8 waves; wave w owns one 16-row M-tile.
// Keys staged by the Tensor Data Mover into double-buffered LDS.
// ---------------------------------------------------------------------------
__global__ __launch_bounds__(256)
void gemm_argmin_kernel(const bf16_t* __restrict__ bhi, const bf16_t* __restrict__ blo,
                        const bf16_t* __restrict__ khi, const bf16_t* __restrict__ klo,
                        const float* __restrict__ k2arr, int* __restrict__ idxout) {
  __shared__ char lds[2][2][NT * 16 * ROWB];   // [buf][hi/lo][64 rows * 144B] = 36KB

  const int tid  = threadIdx.x;
  const int lane = tid & 31;
  const int wv   = tid >> 5;
  const int c    = blockIdx.y;
  const int b0   = (blockIdx.x * 8 + wv) * 16;   // first batch row of this wave's M-tile

  // Make LDS address escape: compiler cannot see the TDM engine's LDS writes.
  {
    unsigned p = (unsigned)(uintptr_t)&lds[0][0][0];
    asm volatile("" :: "s"(p) : "memory");
  }

  // ---- A fragments (batch rows), ISA 16-bit A 16x32 layout, kept in VGPRs ----
  // lane l (<16): M=l, K = kk*32 + {0..7, 16..23}; lane l+16: K = kk*32 + {8..15, 24..31}
  const size_t arow = ((size_t)(b0 + (lane & 15)) * CQ + c) * DQ;
  const int dpart = (lane >> 4) * 8;
  v16bf a_hi[2], a_lo[2];
#pragma unroll
  for (int kk = 0; kk < 2; ++kk) {
    V16U th, tl;
    const bf16_t* pH = bhi + arow + kk * 32 + dpart;
    const bf16_t* pL = blo + arow + kk * 32 + dpart;
    th.h[0] = *(const v8bf*)(pH);
    th.h[1] = *(const v8bf*)(pH + 16);
    tl.h[0] = *(const v8bf*)(pL);
    tl.h[1] = *(const v8bf*)(pL + 16);
    a_hi[kk] = th.v;
    a_lo[kk] = tl.v;
  }

  float runmin[8];
  int   runidx[8];
#pragma unroll
  for (int r = 0; r < 8; ++r) { runmin[r] = __builtin_inff(); runidx[r] = 0; }

  const size_t kbase = (size_t)c * KQ * DQ;
  if (wv == 0) {   // wave-uniform branch: only wave 0 drives the TDM
    tdm_load_tile(khi + kbase, (unsigned)(uintptr_t)&lds[0][0][0]);
    tdm_load_tile(klo + kbase, (unsigned)(uintptr_t)&lds[0][1][0]);
  }

  const int doff = (lane >> 4) * 16;   // B-frag: lanes 16-31 read K = {8..15, 24..31}

  for (int s = 0; s < NSTAGES; ++s) {
    const int buf = s & 1;
    if (wv == 0) {
      if (s + 1 < NSTAGES) {
        const size_t knext = kbase + (size_t)(s + 1) * (NT * 16) * DQ;
        tdm_load_tile(khi + knext, (unsigned)(uintptr_t)&lds[buf ^ 1][0][0]);
        tdm_load_tile(klo + knext, (unsigned)(uintptr_t)&lds[buf ^ 1][1][0]);
        __builtin_amdgcn_s_wait_tensorcnt(2);   // in-order TDM: stage s has landed
      } else {
        __builtin_amdgcn_s_wait_tensorcnt(0);
      }
    }
    __syncthreads();                            // publish TDM-written LDS to all waves

#pragma unroll
    for (int t = 0; t < NT; ++t) {
      const int rrow = t * 16 + (lane & 15);     // LDS key row for this lane's column
      const char* baseH = &lds[buf][0][rrow * ROWB + doff];
      const char* baseL = &lds[buf][1][rrow * ROWB + doff];

      v16bf b_hi[2], b_lo[2];
#pragma unroll
      for (int kk = 0; kk < 2; ++kk) {
        V16U th, tl;
        th.h[0] = *(const v8bf*)(baseH + kk * 64);
        th.h[1] = *(const v8bf*)(baseH + kk * 64 + 32);
        tl.h[0] = *(const v8bf*)(baseL + kk * 64);
        tl.h[1] = *(const v8bf*)(baseL + kk * 64 + 32);
        b_hi[kk] = th.v;
        b_lo[kk] = tl.v;
      }

      // cross = (a_hi+a_lo)(b_hi+b_lo) ~= hi*hi + hi*lo + lo*hi, one accum chain
      v8f acc{};
      acc = __builtin_amdgcn_wmma_f32_16x16x32_bf16(false, a_hi[0], false, b_hi[0], (short)0, acc, false, false);
      acc = __builtin_amdgcn_wmma_f32_16x16x32_bf16(false, a_hi[1], false, b_hi[1], (short)0, acc, false, false);
      acc = __builtin_amdgcn_wmma_f32_16x16x32_bf16(false, a_hi[0], false, b_lo[0], (short)0, acc, false, false);
      acc = __builtin_amdgcn_wmma_f32_16x16x32_bf16(false, a_hi[1], false, b_lo[1], (short)0, acc, false, false);
      acc = __builtin_amdgcn_wmma_f32_16x16x32_bf16(false, a_lo[0], false, b_hi[0], (short)0, acc, false, false);
      acc = __builtin_amdgcn_wmma_f32_16x16x32_bf16(false, a_lo[1], false, b_hi[1], (short)0, acc, false, false);

      // epilogue: score = k2 - 2*cross ; running argmin (strict '<' keeps first hit)
      const int ncol = (s * NT + t) * 16 + (lane & 15);
      const float kv = k2arr[(size_t)c * KQ + ncol];
#pragma unroll
      for (int r = 0; r < 8; ++r) {
        float sc = fmaf(-2.f, acc[r], kv);
        if (sc < runmin[r]) { runmin[r] = sc; runidx[r] = ncol; }
      }
    }
    __syncthreads();                            // all reads of buf done before next TDM overwrite
  }

  // cross-lane argmin: row M=r lives in lanes 0-15 (vgpr r), M=r+8 in lanes 16-31.
  int rid[8];
#pragma unroll
  for (int r = 0; r < 8; ++r) {
    float v = runmin[r];
    int   i = runidx[r];
#pragma unroll
    for (int off = 1; off < 16; off <<= 1) {
      float v2 = __shfl_xor(v, off, 32);
      int   i2 = __shfl_xor(i, off, 32);
      if (v2 < v || (v2 == v && i2 < i)) { v = v2; i = i2; }
    }
    rid[r] = i;
  }
  if (lane == 0) {
#pragma unroll
    for (int r = 0; r < 8; ++r) idxout[(size_t)(b0 + r) * CQ + c] = rid[r];
  } else if (lane == 16) {
#pragma unroll
    for (int r = 0; r < 8; ++r) idxout[(size_t)(b0 + 8 + r) * CQ + c] = rid[r];
  }
}

// ---------------------------------------------------------------------------
// Kernel 4: out[b, v] = mean_c values[c, idx[b,c], v]
// ---------------------------------------------------------------------------
__global__ __launch_bounds__(VQ)
void gather_mean_kernel(const float* __restrict__ values, const int* __restrict__ idx,
                        float* __restrict__ out) {
  __shared__ int sidx[CQ];
  const int b = blockIdx.x, tid = threadIdx.x;
  if (tid < CQ) sidx[tid] = idx[(size_t)b * CQ + tid];
  __syncthreads();
  float acc = 0.f;
#pragma unroll
  for (int cc = 0; cc < CQ; ++cc)
    acc += values[((size_t)cc * KQ + sidx[cc]) * VQ + tid];
  out[(size_t)b * VQ + tid] = acc * (1.0f / CQ);
}

// ---------------------------------------------------------------------------
extern "C" void kernel_launch(void* const* d_in, const int* in_sizes, int n_in,
                              void* d_out, int out_size, void* d_ws, size_t ws_size,
                              hipStream_t stream) {
  (void)in_sizes; (void)n_in; (void)out_size; (void)ws_size;
  const float* batch  = (const float*)d_in[0];   // (B, C, D)
  const float* keys   = (const float*)d_in[1];   // (C, K, D)
  const float* values = (const float*)d_in[2];   // (C, K, V)
  float* out = (float*)d_out;                    // (B, 1, V)

  char* ws = (char*)d_ws;
  size_t o = 0;
  bf16_t* khi = (bf16_t*)(ws + o); o += (size_t)CQ * KQ * DQ * sizeof(bf16_t);  // 8 MB
  bf16_t* klo = (bf16_t*)(ws + o); o += (size_t)CQ * KQ * DQ * sizeof(bf16_t);  // 8 MB
  bf16_t* bhi = (bf16_t*)(ws + o); o += (size_t)BQ * CQ * DQ * sizeof(bf16_t);  // 8 MB
  bf16_t* blo = (bf16_t*)(ws + o); o += (size_t)BQ * CQ * DQ * sizeof(bf16_t);  // 8 MB
  float*  k2  = (float*) (ws + o); o += (size_t)CQ * KQ * sizeof(float);        // 256 KB
  int*    idx = (int*)   (ws + o); o += (size_t)BQ * CQ * sizeof(int);          // 256 KB

  convert_batch_kernel<<<(BQ * CQ * DQ) / (256 * 4), 256, 0, stream>>>(batch, bhi, blo);
  convert_keys_kernel<<<(CQ * KQ) / 256, 256, 0, stream>>>(keys, khi, klo, k2);
  gemm_argmin_kernel<<<dim3(BQ / 128, CQ), 256, 0, stream>>>(bhi, blo, khi, klo, k2, idx);
  gather_mean_kernel<<<BQ, VQ, 0, stream>>>(values, idx, out);
}
